// OnlineDenoisingAutoencoder_18227841204269
// MI455X (gfx1250) — matrix operand order repaired and checked by reading, not compiled, not run on hardware
//
#include <hip/hip_runtime.h>
#include <math.h>

// Problem constants (from reference): INPUT_DIM=1, PROJ=16, HIDDEN=32, B=T=2048
constexpr int kB = 2048;
constexpr int kT = 2048;
constexpr int kH = 32;    // hidden
constexpr int WAVES = 4;  // waves per block (wave32)
constexpr int LROW = 34;  // padded LDS row stride (dwords), even for b64 loads
constexpr int UNR = 4;    // timestep group (one b128 x-load / out-store per group)

typedef __attribute__((ext_vector_type(16))) __bf16        v16bf;
typedef __attribute__((ext_vector_type(8)))  float         v8f;
typedef __attribute__((ext_vector_type(2)))  float         v2f;
typedef __attribute__((ext_vector_type(8)))  unsigned int  v8u;

#if __has_builtin(__builtin_amdgcn_cvt_pk_bf16_f32)
typedef __attribute__((ext_vector_type(2))) __bf16 v2bf;
__device__ __forceinline__ unsigned pk_bf16(float a, float b) {
  v2bf r = __builtin_amdgcn_cvt_pk_bf16_f32(a, b);
  return __builtin_bit_cast(unsigned, r);
}
#else
__device__ __forceinline__ unsigned pk_bf16(float a, float b) {
  // round-to-nearest-even f32 -> bf16, packed pair
  unsigned ua = __float_as_uint(a); ua = (ua + 0x7fffu + ((ua >> 16) & 1u)) >> 16;
  unsigned ub = __float_as_uint(b); ub = (ub + 0x7fffu + ((ub >> 16) & 1u)) >> 16;
  return (ub << 16) | (ua & 0xffffu);
}
#endif

#if __has_builtin(__builtin_amdgcn_tanhf)
__device__ __forceinline__ float ftanh(float x) { return __builtin_amdgcn_tanhf(x); }
#else
__device__ __forceinline__ float ftanh(float x) {
  x = fminf(15.f, fmaxf(-15.f, x));
  float e = __expf(2.f * x);
  return (e - 1.f) / (e + 1.f);
}
#endif
// sigmoid of (2*y) given y = 0.5*x pre-scaled into the matmul weights:
__device__ __forceinline__ float fsig_half(float y) { return fmaf(0.5f, ftanh(y), 0.5f); }

__global__ __launch_bounds__(32 * WAVES)
void lstm_wmma_kernel(const float* __restrict__ x_seq,
                      const float* __restrict__ Wp,  const float* __restrict__ bp,
                      const float* __restrict__ Wih, const float* __restrict__ Whh,
                      const float* __restrict__ bih, const float* __restrict__ bhh,
                      const float* __restrict__ Wo,  const float* __restrict__ bo,
                      float* __restrict__ out)
{
  __shared__ float lds[WAVES * 16 * LROW];
  const int tid  = threadIdx.x;
  const int wave = tid >> 5;
  const int lane = tid & 31;
  const int l16  = lane & 15;
  const int hi   = lane >> 4;                       // 0: lanes 0-15, 1: lanes 16-31
  const int b_base = (blockIdx.x * WAVES + wave) * 16;
  const int row  = b_base + l16;                    // batch row this lane services
  float* wlds = lds + wave * 16 * LROW;

  // ---------------- time-invariant fragments (registers) ----------------
  // Gate tiles: i -> nt 0,1 | f -> nt 2,3 | g -> nt 4,5 | o -> nt 6,7.
  // i/f/o rows pre-scaled by 0.5 so sigmoid is tanh+fma (no per-element mul).
  v2f augB[8];          // f32 B 4x16 frags: v0 = Bv row, v1 = A row (lanes>=16: 0)
  unsigned wbu[8][8];   // bf16 B frags of (scaled) W_hh^T, 32x16 per n-tile
  float2 wo2[8];

  #pragma unroll
  for (int nt = 0; nt < 8; nt++) {
    const float s = (nt == 4 || nt == 5) ? 1.f : 0.5f;   // g-gate unscaled
    int n = nt * 16 + l16;                          // global gate index this lane covers
    float An = 0.f, Bn = 0.f;
    #pragma unroll
    for (int p = 0; p < 16; p++) {                  // A = Wih@Wp, Bv = Wih@bp + bih + bhh
      float w = Wih[n * 16 + p];
      An += w * Wp[p];
      Bn += w * bp[p];
    }
    Bn += bih[n] + bhh[n];
    augB[nt].x = hi ? 0.f : Bn * s;
    augB[nt].y = hi ? 0.f : An * s;
    int koff = hi * 16;
    #pragma unroll
    for (int j = 0; j < 8; j++) {
      float2 wpair = *(const float2*)&Whh[n * kH + ((j >> 2) << 4) + (koff >> 1) + 2 * (j & 3)];
      wbu[nt][j] = pk_bf16(wpair.x * s, wpair.y * s);
    }
  }
  #pragma unroll
  for (int j = 0; j < 8; j++) {                     // Wo pairs matching A-frag K layout
    int kj = ((j >> 2) << 4) + hi * 8 + 2 * (j & 3);
    wo2[j] = *(const float2*)&Wo[kj];
  }
  const float bo0 = bo[0];

  // ---------------- recurrent state ----------------
  v8u hA_u = {};                                    // bf16 A-frag of h (h0 = 0)
  float cst[2][8];                                  // c in C-tile layout (2 k-halves)
  #pragma unroll
  for (int kh = 0; kh < 2; kh++)
    #pragma unroll
    for (int e = 0; e < 8; e++) cst[kh][e] = 0.f;

  const float one_lo = hi ? 0.f : 1.f;

  // double-buffered x: one b128 load per UNR timesteps per lane (contiguous in t)
  float4 xq = *(const float4*)&x_seq[row * kT];

  for (int t0 = 0; t0 < kT; t0 += UNR) {
    const int tn = (t0 + UNR) & (kT - 1);           // wraps to 0 on last group (harmless)
    float4 xq_next = *(const float4*)&x_seq[row * kT + tn];

    float4 outv;
    #pragma unroll
    for (int u = 0; u < UNR; u++) {
      const float xv = (u == 0) ? xq.x : (u == 1) ? xq.y : (u == 2) ? xq.z : xq.w;
      v2f augA; augA.x = one_lo; augA.y = hi ? 0.f : xv;  // A rows = [1, x_m, 0, 0]

      v8f zero = {};
      v8f gt[8];
      v16bf hA = __builtin_bit_cast(v16bf, hA_u);
      #pragma unroll
      for (int nt = 0; nt < 8; nt++) {
        // gates_x: D = [1,x] x [Bv;A]  (f32 16x16x4)
        v8f c0 = __builtin_amdgcn_wmma_f32_16x16x4_f32(
            false, augA, false, augB[nt], (short)0, zero, false, false);
        // + h @ W_hh^T  (bf16 16x16x32, f32 accumulate)
        v16bf wb = __builtin_bit_cast(v16bf, *(v8u*)wbu[nt]);
        gt[nt] = __builtin_amdgcn_wmma_f32_16x16x32_bf16(
            false, hA, false, wb, (short)0, c0, false, false);
      }

      // LSTM cell; i/f/o tiles arrive pre-scaled by 0.5 -> sigmoid = tanh + fma
      #pragma unroll
      for (int kh = 0; kh < 2; kh++) {
        #pragma unroll
        for (int e = 0; e < 8; e++) {
          float iv = fsig_half(gt[0 + kh][e]);
          float fv = fsig_half(gt[2 + kh][e]);
          float gv = ftanh   (gt[4 + kh][e]);
          float ov = fsig_half(gt[6 + kh][e]);
          float c  = fv * cst[kh][e] + iv * gv;
          cst[kh][e] = c;
          float h  = ov * ftanh(c);
          // C-layout: row m = e + hi*8, col k = kh*16 + l16
          wlds[(e + hi * 8) * LROW + kh * 16 + l16] = h;
        }
      }
      asm volatile("s_wait_dscnt 0x0" ::: "memory"); // single-wave LDS RAW fence

      // Transpose h into bf16 A-frag; fold in the f32 output dot.
      float op = 0.f;
      #pragma unroll
      for (int j = 0; j < 8; j++) {
        int kj = ((j >> 2) << 4) + hi * 8 + 2 * (j & 3);
        float2 hp = *(float2*)&wlds[l16 * LROW + kj]; // row m = l16, K pair (kj, kj+1)
        hA_u[j] = pk_bf16(hp.x, hp.y);
        op += hp.x * wo2[j].x + hp.y * wo2[j].y;
      }
      op += __shfl_xor(op, 16);                     // combine complementary K halves
      op += bo0;
      if      (u == 0) outv.x = op;
      else if (u == 1) outv.y = op;
      else if (u == 2) outv.z = op;
      else             outv.w = op;
    }

    if (lane < 16)
      *(float4*)&out[row * kT + t0] = outv;         // coalesced 16B store per group
    xq = xq_next;
  }
}

extern "C" void kernel_launch(void* const* d_in, const int* in_sizes, int n_in,
                              void* d_out, int out_size, void* d_ws, size_t ws_size,
                              hipStream_t stream) {
  const float* x   = (const float*)d_in[0];
  const float* Wp  = (const float*)d_in[1];
  const float* bp  = (const float*)d_in[2];
  const float* Wih = (const float*)d_in[3];
  const float* Whh = (const float*)d_in[4];
  const float* bih = (const float*)d_in[5];
  const float* bhh = (const float*)d_in[6];
  const float* Wo  = (const float*)d_in[7];
  const float* bo  = (const float*)d_in[8];
  float* out = (float*)d_out;

  dim3 grid(kB / (16 * WAVES));   // 32 blocks
  dim3 block(32 * WAVES);         // 4 wave32 waves
  hipLaunchKernelGGL(lstm_wmma_kernel, grid, block, 0, stream,
                     x, Wp, bp, Wih, Whh, bih, bhh, Wo, bo, out);
}
